// SWMHSA_87385404604600
// MI455X (gfx1250) — compile-verified, matrix operand order checked by reading, not run on hardware
//
#include <hip/hip_runtime.h>

typedef __attribute__((ext_vector_type(16))) __bf16 v16bf;
typedef __attribute__((ext_vector_type(8)))  __bf16 v8bf;
typedef __attribute__((ext_vector_type(8)))  float  v8f;

#define HH    60
#define WWID  108
#define CD    512
#define BT    16
#define GWN   14          // window grid width (112/8)
#define NW    112         // windows per batch (8*14)
#define NTOK  6480        // 60*108
#define DH    128
#define WSZ   8

// Load a 16x32 bf16 operand fragment (A-layout; also correct as B operand of
// M^T thanks to the lane-mirrored B layout).  Per lane: two contiguous 8-elem
// (16B) chunks at kbase and kbase+16.
__device__ __forceinline__ v16bf load_frag(const __bf16* base, int row, int kbase, int ld) {
    const __bf16* p = base + row * ld + kbase;
    v8bf lo = *(const v8bf*)(p);
    v8bf hi = *(const v8bf*)(p + 16);
    return __builtin_shufflevector(lo, hi, 0,1,2,3,4,5,6,7,8,9,10,11,12,13,14,15);
}

__global__ void convert_f32_bf16(const float* __restrict__ src, __bf16* __restrict__ dst, int n) {
    int i = (blockIdx.x * blockDim.x + threadIdx.x) * 4;
    if (i + 3 < n) {
        float4 v = *(const float4*)(src + i);
        dst[i + 0] = (__bf16)v.x; dst[i + 1] = (__bf16)v.y;
        dst[i + 2] = (__bf16)v.z; dst[i + 3] = (__bf16)v.w;
    }
}

__global__ void __launch_bounds__(256, 1)
swin_mhsa_fused(const float* __restrict__ x,
                const __bf16* __restrict__ Wqb, const float* __restrict__ bq,
                const __bf16* __restrict__ Wkb, const float* __restrict__ bk,
                const __bf16* __restrict__ Wvb, const float* __restrict__ bv,
                const __bf16* __restrict__ Wob, const float* __restrict__ bo,
                float* __restrict__ out)
{
    extern __shared__ __align__(16) char smem[];
    __bf16* sX  = (__bf16*)smem;        // 64x512 bf16 window input   (64 KB)
    __bf16* sQ  = sX + 64 * CD;         // 64x512 Q                   (64 KB)
    __bf16* sK  = sQ + 64 * CD;         // 64x512 K                   (64 KB)
    __bf16* sVt = sK + 64 * CD;         // 512x64 V^T                 (64 KB)
    float*  sS  = (float*)smem;         // alias sX : 4 heads x 64x64 f32 scores
    __bf16* sP  = sQ;                   // alias sQ : 4 heads x 64x64 bf16 probs
    __bf16* sA  = sK;                   // alias sK : 64x512 bf16 attn output

    const int tid   = threadIdx.x;
    const int lane  = tid & 31;
    const int wv    = tid >> 5;         // wave 0..7
    const int lrow  = lane & 15;
    const int lhalf = lane >> 4;

    const int b  = blockIdx.x / NW;
    const int wd = blockIdx.x % NW;
    const int gh = wd / GWN, gw = wd % GWN;

    // ---- Phase 0: stage input window into LDS as bf16 (zero padding) ----
    {
        const int m   = tid >> 2;              // token 0..63 inside window
        const int seg = (tid & 3) * 128;       // channel segment
        const int row = gh * WSZ + (m >> 3);
        const int col = gw * WSZ + (m & 7);
        __bf16* dst = sX + m * CD + seg;
        if (row < HH && col < WWID) {
            const float4* src = (const float4*)(x + ((size_t)b * NTOK + row * WWID + col) * CD + seg);
            #pragma unroll
            for (int i = 0; i < 16; ++i) {
                float4 a = src[2 * i], c = src[2 * i + 1];
                v8bf o = { (__bf16)a.x, (__bf16)a.y, (__bf16)a.z, (__bf16)a.w,
                           (__bf16)c.x, (__bf16)c.y, (__bf16)c.z, (__bf16)c.w };
                *(v8bf*)(dst + i * 8) = o;
            }
        } else {
            v8bf z = {};
            #pragma unroll
            for (int i = 0; i < 16; ++i) *(v8bf*)(dst + i * 8) = z;
        }
    }
    __syncthreads();

    // ---- Phase 1: Q/K/V projections  C[64x512] = X @ W^T + bias ----
    // One W B-frag shared by 4 independent M-tile accumulators:
    // W read exactly once per WG, and 4-way independent WMMA chains.
    for (int p = 0; p < 3; ++p) {
        const __bf16* Wb = (p == 0) ? Wqb : (p == 1) ? Wkb : Wvb;
        const float*  bb = (p == 0) ? bq  : (p == 1) ? bk  : bv;
        for (int j = 0; j < 4; ++j) {
            const int nt = wv * 4 + j;
            const int n  = nt * 16 + lrow;
            const float bias = bb[n];
            v8f acc[4];
            #pragma unroll
            for (int mt = 0; mt < 4; ++mt)
                #pragma unroll
                for (int r = 0; r < 8; ++r) acc[mt][r] = bias;
            #pragma unroll
            for (int ks = 0; ks < 16; ++ks) {
                v16bf bF = load_frag(Wb, n, ks * 32 + 8 * lhalf, CD);
                #pragma unroll
                for (int mt = 0; mt < 4; ++mt) {
                    v16bf aF = load_frag(sX, mt * 16 + lrow, ks * 32 + 8 * lhalf, CD);
                    acc[mt] = __builtin_amdgcn_wmma_f32_16x16x32_bf16(false, aF, false, bF,
                                                                      (short)0, acc[mt], false, false);
                }
            }
            if (p < 2) {
                __bf16* dst = (p == 0) ? sQ : sK;
                #pragma unroll
                for (int mt = 0; mt < 4; ++mt)
                    #pragma unroll
                    for (int r = 0; r < 8; ++r) {
                        int m = mt * 16 + r + 8 * lhalf;
                        dst[m * CD + n] = (__bf16)acc[mt][r];
                    }
            } else {
                // store V transposed: consecutive accumulator rows -> contiguous LDS
                #pragma unroll
                for (int mt = 0; mt < 4; ++mt)
                    #pragma unroll
                    for (int r = 0; r < 8; ++r) {
                        int m = mt * 16 + r + 8 * lhalf;
                        sVt[n * 64 + m] = (__bf16)acc[mt][r];
                    }
            }
        }
    }
    __syncthreads();

    // ---- Phase 2: S = (Q K^T) * 1/sqrt(dh), per head ----
    // One Q A-frag shared by 4 independent N-tile accumulators.
    {
        const float scale = 0.088388347648318447f; // 1/sqrt(128)
        const int h     = wv >> 1;
        const int mbase = (wv & 1) * 2;
        for (int mi = 0; mi < 2; ++mi) {
            const int mt = mbase + mi;
            v8f acc[4] = {};
            #pragma unroll
            for (int ks = 0; ks < 4; ++ks) {
                v16bf aF = load_frag(sQ, mt * 16 + lrow, h * DH + ks * 32 + 8 * lhalf, CD);
                #pragma unroll
                for (int nt = 0; nt < 4; ++nt) {
                    v16bf bF = load_frag(sK, nt * 16 + lrow, h * DH + ks * 32 + 8 * lhalf, CD);
                    acc[nt] = __builtin_amdgcn_wmma_f32_16x16x32_bf16(false, aF, false, bF,
                                                                      (short)0, acc[nt], false, false);
                }
            }
            #pragma unroll
            for (int nt = 0; nt < 4; ++nt)
                #pragma unroll
                for (int r = 0; r < 8; ++r) {
                    int m = mt * 16 + r + 8 * lhalf;
                    sS[h * 4096 + m * 64 + nt * 16 + lrow] = acc[nt][r] * scale;
                }
        }
    }
    __syncthreads();

    // ---- Softmax over rows (1 thread per (head,row)); write bf16 probs ----
    {
        const int h = tid >> 6, row = tid & 63;
        float* sr = sS + h * 4096 + row * 64;
        float mx = -3.4e38f;
        #pragma unroll
        for (int i = 0; i < 64; ++i) mx = fmaxf(mx, sr[i]);
        float sum = 0.0f;
        #pragma unroll
        for (int i = 0; i < 64; ++i) sum += __expf(sr[i] - mx);
        const float inv = 1.0f / sum;
        __bf16* pr = sP + h * 4096 + row * 64;
        #pragma unroll
        for (int c8 = 0; c8 < 8; ++c8) {
            v8bf o;
            #pragma unroll
            for (int i = 0; i < 8; ++i)
                o[i] = (__bf16)(__expf(sr[c8 * 8 + i] - mx) * inv);
            *(v8bf*)(pr + c8 * 8) = o;
        }
    }
    __syncthreads();

    // ---- Phase 3: attn = P @ V   (one V^T B-frag shared by 4 M-tiles) ----
    for (int j = 0; j < 4; ++j) {
        const int ntG = wv * 4 + j;         // global 16-col tile 0..31
        const int h = ntG >> 3, dt = ntG & 7;
        v8f acc[4] = {};
        #pragma unroll
        for (int ks = 0; ks < 2; ++ks) {
            v16bf bF = load_frag(sVt, h * DH + dt * 16 + lrow, ks * 32 + 8 * lhalf, 64);
            #pragma unroll
            for (int mt = 0; mt < 4; ++mt) {
                v16bf aF = load_frag(sP + h * 4096, mt * 16 + lrow, ks * 32 + 8 * lhalf, 64);
                acc[mt] = __builtin_amdgcn_wmma_f32_16x16x32_bf16(false, aF, false, bF,
                                                                  (short)0, acc[mt], false, false);
            }
        }
        #pragma unroll
        for (int mt = 0; mt < 4; ++mt)
            #pragma unroll
            for (int r = 0; r < 8; ++r) {
                int m = mt * 16 + r + 8 * lhalf;
                sA[m * CD + ntG * 16 + lrow] = (__bf16)acc[mt][r];
            }
    }
    __syncthreads();

    // ---- Phase 4: out = attn @ Wo^T + bo, masked window-merge store ----
    for (int j = 0; j < 4; ++j) {
        const int nt = wv * 4 + j;
        const int n  = nt * 16 + lrow;
        const float bias = bo[n];
        v8f acc[4];
        #pragma unroll
        for (int mt = 0; mt < 4; ++mt)
            #pragma unroll
            for (int r = 0; r < 8; ++r) acc[mt][r] = bias;
        #pragma unroll
        for (int ks = 0; ks < 16; ++ks) {
            v16bf bF = load_frag(Wob, n, ks * 32 + 8 * lhalf, CD);
            #pragma unroll
            for (int mt = 0; mt < 4; ++mt) {
                v16bf aF = load_frag(sA, mt * 16 + lrow, ks * 32 + 8 * lhalf, CD);
                acc[mt] = __builtin_amdgcn_wmma_f32_16x16x32_bf16(false, aF, false, bF,
                                                                  (short)0, acc[mt], false, false);
            }
        }
        #pragma unroll
        for (int mt = 0; mt < 4; ++mt)
            #pragma unroll
            for (int r = 0; r < 8; ++r) {
                int m   = mt * 16 + r + 8 * lhalf;
                int row = gh * WSZ + (m >> 3);
                int col = gw * WSZ + (m & 7);
                if (row < HH && col < WWID)
                    out[((size_t)b * NTOK + row * WWID + col) * CD + n] = acc[mt][r];
            }
    }
}

extern "C" void kernel_launch(void* const* d_in, const int* in_sizes, int n_in,
                              void* d_out, int out_size, void* d_ws, size_t ws_size,
                              hipStream_t stream) {
    const float* x  = (const float*)d_in[0];
    // d_in[1] = t (unused by reference math)
    const float* Wq = (const float*)d_in[2];
    const float* bq = (const float*)d_in[3];
    const float* Wk = (const float*)d_in[4];
    const float* bk = (const float*)d_in[5];
    const float* Wv = (const float*)d_in[6];
    const float* bv = (const float*)d_in[7];
    const float* Wo = (const float*)d_in[8];
    const float* bo = (const float*)d_in[9];

    const int WN = CD * CD; // 262144
    __bf16* wb = (__bf16*)d_ws;
    convert_f32_bf16<<<WN / 1024, 256, 0, stream>>>(Wq, wb + 0 * (size_t)WN, WN);
    convert_f32_bf16<<<WN / 1024, 256, 0, stream>>>(Wk, wb + 1 * (size_t)WN, WN);
    convert_f32_bf16<<<WN / 1024, 256, 0, stream>>>(Wv, wb + 2 * (size_t)WN, WN);
    convert_f32_bf16<<<WN / 1024, 256, 0, stream>>>(Wo, wb + 3 * (size_t)WN, WN);

    const size_t lds = 4 * 64 * CD * sizeof(__bf16); // 256 KB of the WGP's 320 KB
    swin_mhsa_fused<<<BT * NW, 256, lds, stream>>>(
        x,
        wb + 0 * (size_t)WN, bq,
        wb + 1 * (size_t)WN, bk,
        wb + 2 * (size_t)WN, bv,
        wb + 3 * (size_t)WN, bo,
        (float*)d_out);
}
